// SimpleSSM_12876311953912
// MI455X (gfx1250) — compile-verified, matrix operand order checked by reading, not run on hardware
//
#include <hip/hip_runtime.h>

// ---------------------------------------------------------------------------
// SimpleSSM for MI455X (gfx1250, wave32, WMMA).
//   Phase 1: uB[t][b][s] = sum_i u[b,t,i] * Bm[s,i]     (big bf16 WMMA GEMM)
//   Phase 2: sequential scan, x = tanh(x*A^T + uB_t); y = x*C^T
//            4 persistent workgroups (16 batch rows each), A/C resident in LDS.
// ---------------------------------------------------------------------------

typedef __attribute__((ext_vector_type(16))) __bf16 v16bf;
typedef __attribute__((ext_vector_type(8)))  __bf16 v8bf;
typedef __attribute__((ext_vector_type(8)))  float  v8f;

static constexpr int kB    = 64;
static constexpr int kT    = 4096;
static constexpr int kDin  = 128;
static constexpr int kDst  = 256;
static constexpr int kDout = 128;

// 16-bit WMMA A/B fragment layout (ISA 7.12.2): lane half h = lane>>4.
// Fragment halves 0..7  hold K = kbase + 8*h + {0..7}
// Fragment halves 8..15 hold K = kbase + 16 + 8*h + {0..7}
__device__ __forceinline__ v16bf frag_from_f32(const float* __restrict__ row,
                                               int kbase, int h) {
  const float* p0 = row + kbase + h * 8;
  const float* p1 = p0 + 16;
  v16bf f;
#pragma unroll
  for (int j = 0; j < 8; ++j) {
    f[j]     = (__bf16)p0[j];
    f[j + 8] = (__bf16)p1[j];
  }
  return f;
}

__device__ __forceinline__ v16bf frag_from_lds(const __bf16* row, int kbase, int h) {
  const __bf16* p = row + kbase + h * 8;
  v8bf lo = *(const v8bf*)p;          // ds_load_b128
  v8bf hi = *(const v8bf*)(p + 16);   // ds_load_b128
  return __builtin_shufflevector(lo, hi, 0, 1, 2, 3, 4, 5, 6, 7,
                                 8, 9, 10, 11, 12, 13, 14, 15);
}

// ---------------------------------------------------------------------------
// Phase 1: input projection GEMM.  grid = (2048, 4), block = 256 (8 waves).
// Each wave: one 16-row M-tile x four 16-col N-tiles, K = 128 (4 WMMA steps),
// A-fragments held in registers across the 4 N-tiles.
// Output written to workspace in scan-friendly [t][b][s] order.
// ---------------------------------------------------------------------------
__global__ __launch_bounds__(256) void ssm_proj_kernel(
    const float* __restrict__ u, const float* __restrict__ Bm,
    float* __restrict__ uB) {
  const int tid  = threadIdx.x;
  const int w    = tid >> 5;
  const int lane = tid & 31;
  const int h    = lane >> 4;
  const int nl   = lane & 15;

  const int mtile = blockIdx.x * 8 + w;   // 16384 M-tiles total
  const int row0  = mtile * 16;
  const int n0    = blockIdx.y * 64;      // 4 N-tiles per wave

  const float* urow = u + (size_t)(row0 + nl) * kDin;
  v16bf afrag[4];
#pragma unroll
  for (int ks = 0; ks < 4; ++ks) afrag[ks] = frag_from_f32(urow, ks * 32, h);

  v8f acc[4] = {};
#pragma unroll
  for (int nt = 0; nt < 4; ++nt) {
    const int n = n0 + nt * 16 + nl;                 // this lane's output column
    const float* brow = Bm + (size_t)n * kDin;       // Bm row n, contiguous in K
#pragma unroll
    for (int ks = 0; ks < 4; ++ks) {
      v16bf bfrag = frag_from_f32(brow, ks * 32, h);
      acc[nt] = __builtin_amdgcn_wmma_f32_16x16x32_bf16(
          false, afrag[ks], false, bfrag, (short)0, acc[nt], false, false);
    }
  }

  // D layout: lane holds rows m = 8*h + j for column n; scatter to [t][b][s].
#pragma unroll
  for (int nt = 0; nt < 4; ++nt) {
    const int n = n0 + nt * 16 + nl;
#pragma unroll
    for (int j = 0; j < 8; ++j) {
      const int r = row0 + h * 8 + j;   // r = b*T + t
      const int b = r >> 12;            // T = 4096
      const int t = r & (kT - 1);
      uB[((size_t)t * kB + b) * kDst + n] = acc[nt][j];
    }
  }
}

// ---------------------------------------------------------------------------
// Phase 2: recurrent scan.  grid = 4 blocks x 512 threads (16 waves).
// Block handles batch rows [16*blk, 16*blk+16).  A (bf16, 128KB) and C (bf16,
// 64KB) staged in dynamic LDS; state x (16x256 bf16) lives in LDS between
// steps, accumulation & tanh in f32.  Wave w owns state columns s'=16w..16w+15
// for the A-GEMM; waves 0..7 own output columns o=16w..16w+15 for the C-GEMM.
// ---------------------------------------------------------------------------
__global__ __launch_bounds__(512) void ssm_scan_kernel(
    const float* __restrict__ uB, const float* __restrict__ A,
    const float* __restrict__ C, float* __restrict__ out) {
  extern __shared__ __bf16 smem[];
  __bf16* ldsA = smem;                     // 256*256 bf16 = 128 KB
  __bf16* ldsC = smem + kDst * kDst;       // 128*256 bf16 =  64 KB
  __bf16* ldsx = ldsC + kDout * kDst;      //  16*256 bf16 =   8 KB

  const int tid  = threadIdx.x;
  const int w    = tid >> 5;
  const int lane = tid & 31;
  const int h    = lane >> 4;
  const int nl   = lane & 15;
  const int b0   = blockIdx.x * 16;

  for (int i = tid; i < kDst * kDst; i += 512)  ldsA[i] = (__bf16)A[i];
  for (int i = tid; i < kDout * kDst; i += 512) ldsC[i] = (__bf16)C[i];
  for (int i = tid; i < 16 * kDst; i += 512)    ldsx[i] = (__bf16)0.0f;
  __syncthreads();

  // Loop-invariant B-operand fragments of A^T: B[k=s][n=s'] = A[s'][s],
  // i.e. lane reads row (w*16+nl) of A contiguously in K.  Cache in VGPRs.
  const __bf16* arow = ldsA + (size_t)(w * 16 + nl) * kDst;
  v16bf aB[8];
#pragma unroll
  for (int ks = 0; ks < 8; ++ks) aB[ks] = frag_from_lds(arow, ks * 32, h);

  const __bf16* xrow = ldsx + (size_t)nl * kDst;                // A-frag rows
  const __bf16* crow = ldsC + (size_t)(w * 16 + nl) * kDst;     // C-GEMM B-frag

  for (int t = 0; t < kT; ++t) {
    // ---- x_acc = x_{t-1} * A^T for this wave's 16 state columns ----
    v8f acc = {};
#pragma unroll
    for (int ks = 0; ks < 8; ++ks) {
      v16bf xa = frag_from_lds(xrow, ks * 32, h);
      acc = __builtin_amdgcn_wmma_f32_16x16x32_bf16(
          false, xa, false, aB[ks], (short)0, acc, false, false);
    }

    // ---- add input projection, tanh (f32) ----
    const float* uBt = uB + (size_t)t * (kB * kDst) + (size_t)b0 * kDst
                          + (w * 16 + nl);
    float xn[8];
#pragma unroll
    for (int j = 0; j < 8; ++j)
      xn[j] = tanhf(acc[j] + uBt[(h * 8 + j) * kDst]);

    if (t + 1 < kT)   // pull next step's uB slice toward the WGP
      __builtin_prefetch(uBt + kB * kDst + (h * 8) * kDst, 0, 0);

    __syncthreads();                        // all reads of old x done
#pragma unroll
    for (int j = 0; j < 8; ++j)
      ldsx[(h * 8 + j) * kDst + (w * 16 + nl)] = (__bf16)xn[j];
    __syncthreads();                        // new x visible

    // ---- y = x_t * C^T (waves 0..7; wave-uniform branch, EXEC stays full) --
    if (w < 8) {
      v8f yacc = {};
#pragma unroll
      for (int ks = 0; ks < 8; ++ks) {
        v16bf xa = frag_from_lds(xrow, ks * 32, h);
        v16bf cf = frag_from_lds(crow, ks * 32, h);
        yacc = __builtin_amdgcn_wmma_f32_16x16x32_bf16(
            false, xa, false, cf, (short)0, yacc, false, false);
      }
      float* orow = out + (size_t)t * kDout + (w * 16 + nl);
#pragma unroll
      for (int j = 0; j < 8; ++j)
        orow[(size_t)(b0 + h * 8 + j) * (size_t)kT * kDout] = yacc[j];
    }
  }
}

// ---------------------------------------------------------------------------
extern "C" void kernel_launch(void* const* d_in, const int* in_sizes, int n_in,
                              void* d_out, int out_size, void* d_ws,
                              size_t ws_size, hipStream_t stream) {
  (void)in_sizes; (void)n_in; (void)out_size; (void)ws_size;
  const float* u  = (const float*)d_in[0];
  const float* A  = (const float*)d_in[1];
  const float* Bm = (const float*)d_in[2];
  const float* C  = (const float*)d_in[3];
  float* out = (float*)d_out;
  float* uB  = (float*)d_ws;   // kT*kB*kDst f32 = 256 MB staging in [t][b][s]

  ssm_proj_kernel<<<dim3(2048, 4), 256, 0, stream>>>(u, Bm, uB);

  const size_t lds_bytes = (size_t)(kDst * kDst + kDout * kDst + 16 * kDst) * 2;
  ssm_scan_kernel<<<dim3(4), 512, lds_bytes, stream>>>(uB, A, C, out);
}
